// GNN_20693152432864
// MI455X (gfx1250) — matrix operand order repaired and checked by reading, not compiled, not run on hardware
//
#include <hip/hip_runtime.h>
#include <hip/hip_bf16.h>
#include <cstdint>

// ---------------------------------------------------------------------------
// GNN (2-layer GCN with cosine-correlation Laplacian) for MI455X / gfx1250.
// Heavy GEMMs on v_wmma_f32_16x16x32_bf16 (f32 accumulate); tile staging via
// CDNA5 async global->LDS DMA (global_load_async_to_lds_b128 + ASYNCcnt).
// Stage depth BK=64: one barrier/asynccnt-wait per 16 WMMAs per wave.
// ---------------------------------------------------------------------------

typedef __bf16 bf16x16 __attribute__((ext_vector_type(16)));
typedef float  f32x8   __attribute__((ext_vector_type(8)));

#define BM 128
#define BN 128
#define BKS 64          // staged K-depth per LDS buffer (2 WMMA sub-steps)
#define SA (BKS + 8)    // padded LDS row stride (bf16 elems) to spread banks
#define THREADS 256     // 8 waves of 32

union Frag { uint4 u[2]; bf16x16 v; };

// CDNA5 async global->LDS copy, 16B per lane, GVS addressing (SGPR base +
// 32-bit per-lane offset). Tracked by ASYNCcnt.
__device__ __forceinline__ void async_copy16(unsigned lds_off, unsigned voff,
                                             uint64_t sbase)
{
    asm volatile("global_load_async_to_lds_b128 %0, %1, %2"
                 :: "v"(lds_off), "v"(voff), "s"(sbase) : "memory");
}
__device__ __forceinline__ void wait_async0()
{
    asm volatile("s_wait_asynccnt 0" ::: "memory");
}

// NT GEMM core: C[BM x BN] += A[BM x K] * Bsrc[BN x K]^T, both row-major.
// Wave layout: 8 waves as 4 (M) x 2 (N); each wave owns 32x64 = 2x4 frags.
__device__ __forceinline__ void gemm_core(
    const __hip_bfloat16* __restrict__ gA, int lda,
    const __hip_bfloat16* __restrict__ gB, int ldb,
    int K, f32x8 (&acc)[2][4])
{
    __shared__ __hip_bfloat16 As[2][BM][SA];
    __shared__ __hip_bfloat16 Bs[2][BN][SA];

    const int tid   = threadIdx.x;
    const int lane  = tid & 31;
    const int wave  = tid >> 5;
    const int waveM = wave >> 1;   // 0..3
    const int waveN = wave & 1;    // 0..1

#pragma unroll
    for (int i = 0; i < 2; ++i)
#pragma unroll
        for (int j = 0; j < 4; ++j)
#pragma unroll
            for (int r = 0; r < 8; ++r) acc[i][j][r] = 0.0f;

    // staging map: 128 rows x 64 bf16 = 1024 x 16B chunks per matrix,
    // 4 chunks per thread per matrix.
    unsigned offA[4], offB[4], ldsA[4], ldsB[4];
#pragma unroll
    for (int i = 0; i < 4; ++i) {
        const int c  = tid + THREADS * i;
        const int rr = c >> 3;            // 8 chunks per 64-elem row
        const int kc = (c & 7) * 8;
        offA[i] = (unsigned)((rr * lda + kc) * 2);   // global byte offsets
        offB[i] = (unsigned)((rr * ldb + kc) * 2);
        ldsA[i] = (unsigned)(uintptr_t)&As[0][rr][kc]; // LDS byte offsets
        ldsB[i] = (unsigned)(uintptr_t)&Bs[0][rr][kc];
    }
    const unsigned bufA = (unsigned)(BM * SA * sizeof(__hip_bfloat16));
    const unsigned bufB = (unsigned)(BN * SA * sizeof(__hip_bfloat16));

    const int NK = K / BKS;

    auto stage = [&](int s, int kt) {
        const uint64_t bA = (uint64_t)(uintptr_t)(gA + (size_t)kt * BKS);
        const uint64_t bB = (uint64_t)(uintptr_t)(gB + (size_t)kt * BKS);
#pragma unroll
        for (int i = 0; i < 4; ++i)
            async_copy16(ldsA[i] + (unsigned)s * bufA, offA[i], bA);
#pragma unroll
        for (int i = 0; i < 4; ++i)
            async_copy16(ldsB[i] + (unsigned)s * bufB, offB[i], bB);
    };

    stage(0, 0);
    wait_async0();
    __syncthreads();

    for (int kt = 0; kt < NK; ++kt) {
        const int buf = kt & 1;
        if (kt + 1 < NK) stage(buf ^ 1, kt + 1);       // DMA into other buffer
        if (kt + 2 < NK) {                              // L2 prefetch stage kt+2
            __builtin_prefetch(gA + (size_t)(tid >> 3) * lda + (kt + 2) * BKS, 0, 1);
            __builtin_prefetch(gB + (size_t)(tid >> 3) * ldb + (kt + 2) * BKS, 0, 1);
        }

        // two WMMA sub-steps (K=32 each) per staged buffer
#pragma unroll
        for (int s = 0; s < 2; ++s) {
            const int kb = s * 32;
            // A 16x32 bf16: lanes 0-15 -> M=lane, K 0..7 & 16..23; lanes 16-31 -> K 8..15 & 24..31.
            Frag af[2], bfm[4];
            const int klA = (lane & 16) ? 8 : 0;
#pragma unroll
            for (int mf = 0; mf < 2; ++mf) {
                const int row = waveM * 32 + mf * 16 + (lane & 15);
                af[mf].u[0] = *(const uint4*)&As[buf][row][kb + klA];
                af[mf].u[1] = *(const uint4*)&As[buf][row][kb + 16 + klA];
            }
            // B 32x16 bf16: lanes 0-15 -> N=lane, K 0..7 & 16..23; lanes 16-31 -> K 8..15 & 24..31.
            const int klB = (lane & 16) ? 8 : 0;
#pragma unroll
            for (int nf = 0; nf < 4; ++nf) {
                const int col = waveN * 64 + nf * 16 + (lane & 15);
                bfm[nf].u[0] = *(const uint4*)&Bs[buf][col][kb + klB];
                bfm[nf].u[1] = *(const uint4*)&Bs[buf][col][kb + 16 + klB];
            }

#pragma unroll
            for (int mf = 0; mf < 2; ++mf)
#pragma unroll
                for (int nf = 0; nf < 4; ++nf)
                    acc[mf][nf] = __builtin_amdgcn_wmma_f32_16x16x32_bf16(
                        false, af[mf].v, false, bfm[nf].v,
                        (short)0, acc[mf][nf], false, false);
        }

        if (kt + 1 < NK) {
            wait_async0();       // this wave's DMA into buf^1 done
            __syncthreads();     // every wave's DMA done -> safe to read buf^1
        }
    }
}

// C/D f32 16x16 layout: VGPR r, lanes 0-15 -> (M=r, N=lane); lanes 16-31 -> (M=8+r, N=lane-16).

// --- Gram + adjacency: dot = Xbf @ Xbf^T; adj = (dot > thr*ni*nj); deg += rowsum
__global__ __launch_bounds__(THREADS)
void gram_kernel(const __hip_bfloat16* __restrict__ xbf,
                 const float* __restrict__ norm,
                 __hip_bfloat16* __restrict__ adj,
                 float* __restrict__ deg, int C, int F)
{
    const int b  = blockIdx.z;
    const int m0 = blockIdx.y * BM;
    const int n0 = blockIdx.x * BN;
    const __hip_bfloat16* gA = xbf + (size_t)b * C * F + (size_t)m0 * F;
    const __hip_bfloat16* gB = xbf + (size_t)b * C * F + (size_t)n0 * F;
    f32x8 acc[2][4];
    gemm_core(gA, F, gB, F, F, acc);

    const int lane  = threadIdx.x & 31;
    const int wave  = threadIdx.x >> 5;
    const int waveM = wave >> 1, waveN = wave & 1;
    const int half  = (lane & 16) ? 8 : 0;
    const float* normB = norm + (size_t)b * C;
    float* degB        = deg  + (size_t)b * C;
    __hip_bfloat16* adjB = adj + (size_t)b * C * C;

    // per-lane column norms: depend only on nf, hoist out of mf/r loops
    float njv[4];
#pragma unroll
    for (int nf = 0; nf < 4; ++nf)
        njv[nf] = normB[n0 + waveN * 64 + nf * 16 + (lane & 15)];

#pragma unroll
    for (int mf = 0; mf < 2; ++mf) {
#pragma unroll
        for (int r = 0; r < 8; ++r) {
            const int i = m0 + waveM * 32 + mf * 16 + half + r;
            const float tni = 0.005f * normB[i];   // thr * ni  (ni,nj > 0)
            float rowsum = 0.0f;
#pragma unroll
            for (int nf = 0; nf < 4; ++nf) {
                const int j = n0 + waveN * 64 + nf * 16 + (lane & 15);
                // dot/(ni*nj) > thr  <=>  dot > thr*ni*nj   (no division)
                const float a = (acc[mf][nf][r] > tni * njv[nf]) ? 1.0f : 0.0f;
                adjB[(size_t)i * C + j] = __float2bfloat16(a);
                rowsum += a;
            }
            rowsum += __shfl_xor(rowsum, 1);
            rowsum += __shfl_xor(rowsum, 2);
            rowsum += __shfl_xor(rowsum, 4);
            rowsum += __shfl_xor(rowsum, 8);
            if ((lane & 15) == 0) atomicAdd(&degB[i], rowsum);  // exact int sums
        }
    }
}

// --- Propagation: lx[i][f] = dis_i * sum_j adj[i][j] * hsT[f][j]
__global__ __launch_bounds__(THREADS)
void prop_kernel(const __hip_bfloat16* __restrict__ adj,
                 const __hip_bfloat16* __restrict__ hsT,
                 const float* __restrict__ dis,
                 __hip_bfloat16* __restrict__ lx, int C, int F)
{
    const int b  = blockIdx.z;
    const int m0 = blockIdx.y * BM;
    const int n0 = blockIdx.x * BN;
    const __hip_bfloat16* gA = adj + (size_t)b * C * C + (size_t)m0 * C;
    const __hip_bfloat16* gB = hsT + (size_t)b * F * C + (size_t)n0 * C;
    f32x8 acc[2][4];
    gemm_core(gA, C, gB, C, C, acc);

    const int lane  = threadIdx.x & 31;
    const int wave  = threadIdx.x >> 5;
    const int waveM = wave >> 1, waveN = wave & 1;
    const int half  = (lane & 16) ? 8 : 0;

#pragma unroll
    for (int mf = 0; mf < 2; ++mf) {
#pragma unroll
        for (int r = 0; r < 8; ++r) {
            const int i = m0 + waveM * 32 + mf * 16 + half + r;
            const float di = dis[(size_t)b * C + i];
#pragma unroll
            for (int nf = 0; nf < 4; ++nf) {
                const int f = n0 + waveN * 64 + nf * 16 + (lane & 15);
                lx[((size_t)b * C + i) * F + f] =
                    __float2bfloat16(acc[mf][nf][r] * di);
            }
        }
    }
}

// --- Feature: h = relu(lx @ W + b).  MODE 0: write fp32 d_out.
//     MODE 1: write next layer's hsT[g][i] = dis_i * h[i][g]  (bf16, transposed)
template <int MODE>
__global__ __launch_bounds__(THREADS)
void feat_kernel(const __hip_bfloat16* __restrict__ lxin,
                 const __hip_bfloat16* __restrict__ WT,
                 const float* __restrict__ dis,
                 const float* __restrict__ biasPtr,
                 float* __restrict__ outF,
                 __hip_bfloat16* __restrict__ outT,
                 int C, int F)
{
    const int b  = blockIdx.z;
    const int m0 = blockIdx.y * BM;
    const int n0 = blockIdx.x * BN;
    const __hip_bfloat16* gA = lxin + (size_t)b * C * F + (size_t)m0 * F;
    const __hip_bfloat16* gB = WT + (size_t)n0 * F;
    f32x8 acc[2][4];
    gemm_core(gA, F, gB, F, F, acc);

    const float bias = biasPtr[0];
    const int lane  = threadIdx.x & 31;
    const int wave  = threadIdx.x >> 5;
    const int waveM = wave >> 1, waveN = wave & 1;
    const int half  = (lane & 16) ? 8 : 0;

#pragma unroll
    for (int mf = 0; mf < 2; ++mf) {
#pragma unroll
        for (int r = 0; r < 8; ++r) {
            const int i = m0 + waveM * 32 + mf * 16 + half + r;
            const float di = dis[(size_t)b * C + i];
#pragma unroll
            for (int nf = 0; nf < 4; ++nf) {
                const int g = n0 + waveN * 64 + nf * 16 + (lane & 15);
                const float v = fmaxf(acc[mf][nf][r] + bias, 0.0f);
                if (MODE == 0)
                    outF[((size_t)b * C + i) * F + g] = v;
                else
                    outT[((size_t)b * F + g) * C + i] = __float2bfloat16(v * di);
            }
        }
    }
}

// --- Prep: bf16 copy of x + per-row L2 norms
__global__ __launch_bounds__(256)
void prep_x_kernel(const float* __restrict__ x,
                   __hip_bfloat16* __restrict__ xbf,
                   float* __restrict__ norm, int F)
{
    const int bc = blockIdx.x;
    const float* row = x + (size_t)bc * F;
    __hip_bfloat16* orow = xbf + (size_t)bc * F;
    float s = 0.0f;
    for (int f = threadIdx.x; f < F; f += blockDim.x) {
        const float v = row[f];
        s += v * v;
        orow[f] = __float2bfloat16(v);
    }
    s += __shfl_xor(s, 1);
    s += __shfl_xor(s, 2);
    s += __shfl_xor(s, 4);
    s += __shfl_xor(s, 8);
    s += __shfl_xor(s, 16);
    __shared__ float wsum[8];
    if ((threadIdx.x & 31) == 0) wsum[threadIdx.x >> 5] = s;
    __syncthreads();
    if (threadIdx.x == 0) {
        float t = 0.0f;
        for (int w = 0; w < 8; ++w) t += wsum[w];
        norm[bc] = sqrtf(t);
    }
}

// --- Prep: WT[g][f] = bf16(W[f][g])  (32x32 LDS transpose)
__global__ void prep_w_kernel(const float* __restrict__ W,
                              __hip_bfloat16* __restrict__ WT, int F)
{
    __shared__ float tile[32][33];
    const int f0 = blockIdx.y * 32, g0 = blockIdx.x * 32;
    const int tx = threadIdx.x, ty = threadIdx.y;
#pragma unroll
    for (int i = 0; i < 4; ++i)
        tile[ty + i * 8][tx] = W[(size_t)(f0 + ty + i * 8) * F + g0 + tx];
    __syncthreads();
#pragma unroll
    for (int i = 0; i < 4; ++i)
        WT[(size_t)(g0 + ty + i * 8) * F + f0 + tx] =
            __float2bfloat16(tile[tx][ty + i * 8]);
}

// --- dis = 1/sqrt(deg)
__global__ void dis_kernel(const float* __restrict__ deg,
                           float* __restrict__ dis, int n)
{
    const int i = blockIdx.x * blockDim.x + threadIdx.x;
    if (i < n) dis[i] = rsqrtf(fmaxf(deg[i], 1.0f));
}

// --- hsT[b][f][j] = bf16(dis_j * x[b][j][f])  (scaled 32x32 transpose)
__global__ void build_hsT_kernel(const float* __restrict__ x,
                                 const float* __restrict__ dis,
                                 __hip_bfloat16* __restrict__ hsT, int C, int F)
{
    __shared__ float tile[32][33];
    const int b  = blockIdx.z;
    const int j0 = blockIdx.y * 32, f0 = blockIdx.x * 32;
    const int tx = threadIdx.x, ty = threadIdx.y;
#pragma unroll
    for (int i = 0; i < 4; ++i) {
        const int j = j0 + ty + i * 8;
        tile[ty + i * 8][tx] =
            x[((size_t)b * C + j) * F + f0 + tx] * dis[(size_t)b * C + j];
    }
    __syncthreads();
#pragma unroll
    for (int i = 0; i < 4; ++i) {
        const int f = f0 + ty + i * 8;
        hsT[((size_t)b * F + f) * C + j0 + tx] =
            __float2bfloat16(tile[tx][ty + i * 8]);
    }
}

extern "C" void kernel_launch(void* const* d_in, const int* in_sizes, int n_in,
                              void* d_out, int out_size, void* d_ws, size_t ws_size,
                              hipStream_t stream)
{
    (void)in_sizes; (void)n_in; (void)out_size; (void)ws_size;
    const float* x  = (const float*)d_in[0];
    const float* W1 = (const float*)d_in[1];
    const float* b1 = (const float*)d_in[2];
    const float* W2 = (const float*)d_in[3];
    const float* b2 = (const float*)d_in[4];
    float* out = (float*)d_out;
    const int B = 8, C = 2048, F = 512;

    char* ws = (char*)d_ws;
    size_t off = 0;
    auto alloc = [&](size_t bytes) -> void* {
        void* p = ws + off;
        off = (off + bytes + 255) & ~(size_t)255;
        return p;
    };
    __hip_bfloat16* xbf = (__hip_bfloat16*)alloc((size_t)B * C * F * 2);
    __hip_bfloat16* adj = (__hip_bfloat16*)alloc((size_t)B * C * C * 2);
    __hip_bfloat16* hsT = (__hip_bfloat16*)alloc((size_t)B * F * C * 2);
    __hip_bfloat16* lx  = (__hip_bfloat16*)alloc((size_t)B * C * F * 2);
    __hip_bfloat16* WT1 = (__hip_bfloat16*)alloc((size_t)F * F * 2);
    __hip_bfloat16* WT2 = (__hip_bfloat16*)alloc((size_t)F * F * 2);
    float* norm = (float*)alloc((size_t)B * C * 4);
    float* deg  = (float*)alloc((size_t)B * C * 4);
    float* dis  = (float*)alloc((size_t)B * C * 4);

    prep_x_kernel<<<B * C, 256, 0, stream>>>(x, xbf, norm, F);
    prep_w_kernel<<<dim3(F / 32, F / 32), dim3(32, 8), 0, stream>>>(W1, WT1, F);
    prep_w_kernel<<<dim3(F / 32, F / 32), dim3(32, 8), 0, stream>>>(W2, WT2, F);
    hipMemsetAsync(deg, 0, (size_t)B * C * 4, stream);

    gram_kernel<<<dim3(C / BN, C / BM, B), THREADS, 0, stream>>>(xbf, norm, adj, deg, C, F);
    dis_kernel<<<(B * C + 255) / 256, 256, 0, stream>>>(deg, dis, B * C);
    build_hsT_kernel<<<dim3(F / 32, C / 32, B), dim3(32, 8), 0, stream>>>(x, dis, hsT, C, F);

    // layer 1
    prop_kernel<<<dim3(F / BN, C / BM, B), THREADS, 0, stream>>>(adj, hsT, dis, lx, C, F);
    feat_kernel<1><<<dim3(F / BN, C / BM, B), THREADS, 0, stream>>>(lx, WT1, dis, b1, nullptr, hsT, C, F);
    // layer 2
    prop_kernel<<<dim3(F / BN, C / BM, B), THREADS, 0, stream>>>(adj, hsT, dis, lx, C, F);
    feat_kernel<0><<<dim3(F / BN, C / BM, B), THREADS, 0, stream>>>(lx, WT2, dis, b2, out, nullptr, C, F);
}
